// edgeperception_59407987638781
// MI455X (gfx1250) — compile-verified, matrix operand order checked by reading, not run on hardware
//
#include <hip/hip_runtime.h>
#include <hip/hip_bf16.h>

// Problem constants (from setup_inputs: x is (2,128,96,96) f32)
#define B_   2
#define C_   128
#define H_   96
#define W_   96
#define HW_  (H_*W_)
#define NPOSG (B_*HW_)
#define EPSV 1e-8f

// Tiling for the WMMA Gram kernel
#define TILE 16     // centers per workgroup (one wave)
#define HALO 4      // 9x9 window
#define ROWS 9      // dy in [-4,4]
#define COLS 32     // staged columns: w0-4 .. w0+27 (covers j=0..23 + pad)
#define CSTR 136    // bf16 elems per staged position (128 + 8 pad, 272B = 16B aligned)
#define DSTR 33     // ldots row stride (floats): halves land on disjoint LDS banks

typedef __attribute__((ext_vector_type(16))) __bf16 v16bf;
typedef __attribute__((ext_vector_type(8)))  float  v8f;

__device__ __forceinline__ unsigned bf16_rne(float f) {
    unsigned u = __float_as_uint(f);
    return ((u + 0x7FFFu + ((u >> 16) & 1u)) >> 16) & 0xFFFFu;
}

// ---------------- Kernel 1: per-position L2 norms ----------------
// grid (1, H, B), block 96 (w = tid): coalesced along w, no div/mod.
__global__ void k_norm(const float* __restrict__ x, float* __restrict__ nrm) {
    const int w = threadIdx.x;
    const int h = blockIdx.y;
    const int b = blockIdx.z;
    const float* xp = x + (size_t)b * C_ * HW_ + (size_t)h * W_ + w;
    float acc = 0.f;
#pragma unroll 8
    for (int c = 0; c < C_; ++c) {
        float v = xp[(size_t)c * HW_];
        acc += v * v;
    }
    nrm[((size_t)b * H_ + h) * W_ + w] = sqrtf(acc);
}

// ---------------- Kernel 2: 9x9 Gram via WMMA, large count -> s, store 5x5 dots ----------------
__global__ __launch_bounds__(32) void k_sim_large(
    const float* __restrict__ x, const float* __restrict__ nrm,
    const float* __restrict__ th1p, float* __restrict__ sbuf,
    float* __restrict__ dots)
{
    extern __shared__ unsigned char smem[];
    unsigned short* tile  = (unsigned short*)smem;                       // 288*136*2 = 78336 B
    float*          lnrm  = (float*)(smem + (size_t)ROWS * COLS * CSTR * 2); // 288 floats
    float*          ldots = lnrm + ROWS * COLS;                          // 80*33 floats = 10560 B

    const int lane = threadIdx.x;        // 0..31
    const int half = lane >> 4;          // 0: rows M=0..7, 1: rows M=8..15
    const int lid  = lane & 15;          // output column N within tile
    const int w0   = blockIdx.x * TILE;
    const int h    = blockIdx.y;
    const int b    = blockIdx.z;
    const float th1 = *th1p;

    const float* xb = x + (size_t)b * C_ * HW_;

    // ---- stage 9x32 halo of bf16 feature vectors + norms into LDS ----
    {
        const int m  = lane;                 // staged column 0..31
        const int ww = w0 - HALO + m;
        const bool vcol = (ww >= 0) && (ww < W_);
        for (int r = 0; r < ROWS; ++r) {
            const int hh = h + r - HALO;
            const bool v = vcol && (hh >= 0) && (hh < H_);
            const int pos = r * COLS + m;
            lnrm[pos] = v ? nrm[((size_t)b * H_ + hh) * W_ + ww] : 0.f;
            const float* pg = xb + (size_t)hh * W_ + ww;  // +c*HW per channel
            if (v) __builtin_prefetch(pg, 0, 1);          // global_prefetch_b8
            for (int c = 0; c < C_; c += 4) {
                float f0 = 0.f, f1 = 0.f, f2 = 0.f, f3 = 0.f;
                if (v) {
                    const float* q = pg + (size_t)c * HW_;
                    f0 = q[0]; f1 = q[HW_]; f2 = q[2 * HW_]; f3 = q[3 * HW_];
                }
                unsigned lo = bf16_rne(f0) | (bf16_rne(f1) << 16);
                unsigned hi = bf16_rne(f2) | (bf16_rne(f3) << 16);
                unsigned long long pk = ((unsigned long long)hi << 32) | lo;
                *(unsigned long long*)(tile + pos * CSTR + c) = pk;  // 8B aligned
            }
        }
    }
    __syncthreads();

    // ---- fragment loader: matches 16-bit A/B 16x32 VGPR layout ----
    // lanes 0-15 hold K {0..7,16..23}, lanes 16-31 hold K {8..15,24..31} of each 32-chunk
    auto ldfrag = [&](int pos, int q) -> v16bf {
        union { v16bf v; uint4 u[2]; } fr;
        const unsigned short* p = tile + pos * CSTR + q * 32 + half * 8;
        fr.u[0] = *(const uint4*)(p);        // K base+0..7   (16B, aligned)
        fr.u[1] = *(const uint4*)(p + 16);   // K base+16..23 (16B, aligned)
        return fr.v;
    };

    // A fragments: centers, row dy=0 (r=4), columns w0..w0+15 (staged m = 4..19)
    v16bf afr[4];
    const int aPos = 4 * COLS + HALO + lid;
#pragma unroll
    for (int q = 0; q < 4; ++q) afr[q] = ldfrag(aPos, q);

    // center norms for this lane's 8 rows (register-resident, static-indexed)
    float ncv[8];
#pragma unroll
    for (int e = 0; e < 8; ++e) ncv[e] = lnrm[4 * COLS + HALO + half * 8 + e];

    int cnt[8] = {0, 0, 0, 0, 0, 0, 0, 0};
    const size_t rowbase = ((size_t)b * H_ + h) * W_ + w0;   // global pos of center 0

#pragma unroll
    for (int r = 0; r < ROWS; ++r) {
#pragma unroll
        for (int t = 0; t < 2; ++t) {
            const int j    = t * 16 + lid;        // neighbor column index (0..31)
            const int bPos = r * COLS + j;
            v8f acc = {0.f, 0.f, 0.f, 0.f, 0.f, 0.f, 0.f, 0.f};
#pragma unroll
            for (int q = 0; q < 4; ++q) {
                v16bf bfr = ldfrag(bPos, q);
                acc = __builtin_amdgcn_wmma_f32_16x16x32_bf16(
                    false, afr[q], false, bfr, (short)0, acc, false, false);
            }
            const float npv = lnrm[bPos];
#pragma unroll
            for (int e = 0; e < 8; ++e) {
                const int   M   = half * 8 + e;   // center index in tile
                const int   rel = j - M;          // dx + 4
                const float d   = acc[e];
                // branch-free large-window count (9x9; center contributes 0)
                const float thr = th1 * fmaxf(ncv[e] * npv, EPSV);
                cnt[e] += (((unsigned)rel <= 8u) && (d < thr)) ? 1 : 0;
                // central 5 rows: dump unconditionally to LDS (r is compile-time here)
                if (r >= 2 && r <= 6) {
                    ldots[(M * 5 + (r - 2)) * DSTR + j] = d;
                }
            }
        }
    }

    // ---- reduce counts across the 16 lanes of each half, emit s = 1+sqrt(count) ----
#pragma unroll
    for (int e = 0; e < 8; ++e) {
        int c = cnt[e];
        c += __shfl_xor(c, 1, 32);
        c += __shfl_xor(c, 2, 32);
        c += __shfl_xor(c, 4, 32);
        c += __shfl_xor(c, 8, 32);
        cnt[e] = c;
    }
    if (lid == 0) {
#pragma unroll
        for (int e = 0; e < 8; ++e)
            sbuf[rowbase + half * 8 + e] = 1.0f + sqrtf((float)cnt[e]);
    }

    __syncthreads();

    // ---- coalesced writeout of the 16x25 central dots: dots[rowbase*25 + f] ----
    {
        float* gdots = dots + rowbase * 25;
        for (int base = 0; base < TILE * 25; base += 32) {
            const int f = base + lane;
            if (f < TILE * 25) {
                const int M   = f / 25;
                const int idx = f - M * 25;
                const int r2  = idx / 5;
                const int dxi = idx - r2 * 5;       // rel-2
                const int jj  = M + dxi + 2;
                gdots[f] = ldots[(M * 5 + r2) * DSTR + jj];
            }
        }
    }
}

// ---------------- Kernel 3: second-pass predicate on scaled dots, write output ----------------
// grid (1, H, B), block 96.
__global__ void k_small(const float* __restrict__ x, const float* __restrict__ nrm,
                        const float* __restrict__ sbuf, const float* __restrict__ dots,
                        const float* __restrict__ th2p, float* __restrict__ out)
{
    const int w = threadIdx.x;
    const int h = blockIdx.y;
    const int b = blockIdx.z;
    const size_t p = ((size_t)b * H_ + h) * W_ + w;
    const float th2 = *th2p;
    const float sc  = sbuf[p];
    const float ncv = nrm[p];
    int count = 0;
#pragma unroll
    for (int dy = -2; dy <= 2; ++dy) {
#pragma unroll
        for (int dx = -2; dx <= 2; ++dx) {
            const float d = dots[p * 25 + (dy + 2) * 5 + (dx + 2)];
            const int hh = h + dy, ww = w + dx;
            float npv = 0.f, sp = 1.f;
            if (hh >= 0 && hh < H_ && ww >= 0 && ww < W_) {
                const size_t q = ((size_t)b * H_ + hh) * W_ + ww;
                npv = nrm[q]; sp = sbuf[q];
            }
            // x_mask = (1+large_mask)*x is a positive per-position scaling:
            // dot' = sc*sp*dot, n_c' = sc*ncv, n_p' = sp*npv  (exact reference semantics)
            const float scale = sc * sp;
            count += (scale * d < th2 * fmaxf(scale * ncv * npv, EPSV)) ? 1 : 0;
        }
    }
    const float sm = 1.0f + sqrtf((float)count);
    const float* xp = x + (size_t)b * C_ * HW_ + (size_t)h * W_ + w;
    float*       op = out + (size_t)b * C_ * HW_ + (size_t)h * W_ + w;
#pragma unroll 8
    for (int c = 0; c < C_; ++c)
        op[(size_t)c * HW_] = xp[(size_t)c * HW_] * sm;
}

extern "C" void kernel_launch(void* const* d_in, const int* in_sizes, int n_in,
                              void* d_out, int out_size, void* d_ws, size_t ws_size,
                              hipStream_t stream) {
    const float* x   = (const float*)d_in[0];
    const float* th1 = (const float*)d_in[1];
    const float* th2 = (const float*)d_in[2];
    float* out = (float*)d_out;

    // workspace: norms (18432) | s (18432) | 5x5 dots (18432*25)  ~ 1.99 MB
    float* nrm  = (float*)d_ws;
    float* sbuf = nrm + NPOSG;
    float* dots = sbuf + NPOSG;

    dim3 gpos(1, H_, B_);
    k_norm<<<gpos, 96, 0, stream>>>(x, nrm);

    dim3 g2(W_ / TILE, H_, B_);   // (6, 96, 2) workgroups, 1 wave each
    size_t smem = (size_t)ROWS * COLS * CSTR * 2      // bf16 feature tile
                + (size_t)ROWS * COLS * 4             // norms
                + (size_t)TILE * 5 * DSTR * 4;        // central dot staging
    k_sim_large<<<g2, 32, smem, stream>>>(x, nrm, th1, sbuf, dots);

    k_small<<<gpos, 96, 0, stream>>>(x, nrm, sbuf, dots, th2, out);
}
